// TopMAttention_5686536700002
// MI455X (gfx1250) — compile-verified, matrix operand order checked by reading, not run on hardware
//
#include <hip/hip_runtime.h>

typedef unsigned short u16;
typedef unsigned int u32;

typedef __attribute__((ext_vector_type(16))) __bf16 v16bf;
typedef __attribute__((ext_vector_type(8)))  float  v8f;
typedef __attribute__((ext_vector_type(4)))  int    v4i;

struct alignas(16) F4 { float x, y, z, w; };
struct alignas(16) U4 { u32 x, y, z, w; };

union Frag { U4 q[2]; v16bf v; };

#define NB     2
#define S_LEN  2048
#define E_DIM  1024
#define H_NUM  16
#define D_DIM  64
#define M_TOP  64
#define R_ROWS (NB * S_LEN)   // 4096

// ---------- bf16 helpers (hardware cvt) ----------
__device__ inline u16 f2bf(float f) {
    union { __bf16 b; u16 u; } c; c.b = (__bf16)f; return c.u;
}
__device__ inline u32 packbf(float lo, float hi) {
    union { __bf16 b[2]; u32 u; } c;
    c.b[0] = (__bf16)lo; c.b[1] = (__bf16)hi; return c.u;
}
__device__ inline v8f wmma_bf16(const Frag& a, const Frag& b, v8f c) {
    return __builtin_amdgcn_wmma_f32_16x16x32_bf16(
        false, a.v, false, b.v, (short)0, c, false, false);
}

// ---------- async global->LDS staging (CDNA5 path, guarded) ----------
#if defined(__has_builtin)
#if __has_builtin(__builtin_amdgcn_global_load_async_to_lds_b128)
#define HAS_ASYNC_LDS 1
#endif
#endif
#ifndef HAS_ASYNC_LDS
#define HAS_ASYNC_LDS 0
#endif

typedef __attribute__((address_space(1))) v4i v4i_gbl;   // global int4
typedef __attribute__((address_space(3))) v4i v4i_lds;   // LDS int4

// copy 16 bytes global -> LDS (per active lane)
__device__ inline void stage16(const u16* __restrict__ g, u16* l) {
#if HAS_ASYNC_LDS
    __builtin_amdgcn_global_load_async_to_lds_b128(
        (v4i_gbl*)(unsigned long long)g,
        (v4i_lds*)(unsigned int)(unsigned long long)l, 0, 0);
#else
    *(U4*)l = *(const U4*)g;
#endif
}
__device__ inline void stage_wait() {
#if HAS_ASYNC_LDS
    asm volatile("s_wait_asynccnt 0x0" ::: "memory");
#endif
}

// ---------- weight f32 -> bf16 ----------
__global__ __launch_bounds__(256) void k_cvt_weights(
    const float* __restrict__ wv, const float* __restrict__ wk,
    const float* __restrict__ wq, const float* __restrict__ wo,
    u16* __restrict__ WvB, u16* __restrict__ WkB,
    u16* __restrict__ WqB, u16* __restrict__ WoB)
{
    const int n = E_DIM * E_DIM;
    for (int i = blockIdx.x * blockDim.x + threadIdx.x; i < n;
         i += gridDim.x * blockDim.x) {
        WvB[i] = f2bf(wv[i]);
        WkB[i] = f2bf(wk[i]);
        WqB[i] = f2bf(wq[i]);
        WoB[i] = f2bf(wo[i]);
    }
}

// ---------- GEMM: C[m][n] = sum_k A_f32[m][k] * Wbf[n][k] ----------
// mode==0: write bf16 to Cb; mode==1: write f32 to Cf.
__global__ __launch_bounds__(256) void k_gemm_a32(
    const float* __restrict__ A, const u16* __restrict__ Wb,
    float* __restrict__ Cf, u16* __restrict__ Cb, int mode)
{
    __shared__ alignas(16) u16 Al[64 * 72];
    __shared__ alignas(16) u16 Bl[64 * 72];

    const int t    = threadIdx.x;
    const int wv_  = t >> 5, lane = t & 31;
    const int mi   = wv_ & 3, nh = wv_ >> 2;      // 4 m-tiles, 2 n-groups
    const int m_l  = lane & 15, h2 = lane >> 4;
    const int m0   = blockIdx.y * 64, n0 = blockIdx.x * 64;
    const int ar   = t >> 2;                       // 0..63
    const int ac   = (t & 3) * 16;                 // 0,16,32,48

    v8f acc[2];
    acc[0] = {}; acc[1] = {};

    for (int k = 0; k < E_DIM; k += 64) {
        // stage A (f32 -> bf16 in registers), B via async copy
        {
            const float* src = A + (size_t)(m0 + ar) * E_DIM + k + ac;
            F4 f0 = *(const F4*)(src + 0);
            F4 f1 = *(const F4*)(src + 4);
            F4 f2 = *(const F4*)(src + 8);
            F4 f3 = *(const F4*)(src + 12);
            U4 u0, u1;
            u0.x = packbf(f0.x, f0.y); u0.y = packbf(f0.z, f0.w);
            u0.z = packbf(f1.x, f1.y); u0.w = packbf(f1.z, f1.w);
            u1.x = packbf(f2.x, f2.y); u1.y = packbf(f2.z, f2.w);
            u1.z = packbf(f3.x, f3.y); u1.w = packbf(f3.z, f3.w);
            *(U4*)&Al[ar * 72 + ac]     = u0;
            *(U4*)&Al[ar * 72 + ac + 8] = u1;
            const u16* bs = Wb + (size_t)(n0 + ar) * E_DIM + k + ac;
            stage16(bs,     &Bl[ar * 72 + ac]);
            stage16(bs + 8, &Bl[ar * 72 + ac + 8]);
        }
        stage_wait();
        __syncthreads();

        Frag a0, a1;
        const int arow = (mi * 16 + m_l) * 72;
        a0.q[0] = *(const U4*)&Al[arow + h2 * 8];
        a0.q[1] = *(const U4*)&Al[arow + 16 + h2 * 8];
        a1.q[0] = *(const U4*)&Al[arow + 32 + h2 * 8];
        a1.q[1] = *(const U4*)&Al[arow + 48 + h2 * 8];

        const int br0 = (nh * 16 + m_l) * 72;
        const int br1 = ((nh + 2) * 16 + m_l) * 72;
        Frag b00, b01, b10, b11;
        b00.q[0] = *(const U4*)&Bl[br0 + h2 * 8];
        b00.q[1] = *(const U4*)&Bl[br0 + 16 + h2 * 8];
        b01.q[0] = *(const U4*)&Bl[br0 + 32 + h2 * 8];
        b01.q[1] = *(const U4*)&Bl[br0 + 48 + h2 * 8];
        b10.q[0] = *(const U4*)&Bl[br1 + h2 * 8];
        b10.q[1] = *(const U4*)&Bl[br1 + 16 + h2 * 8];
        b11.q[0] = *(const U4*)&Bl[br1 + 32 + h2 * 8];
        b11.q[1] = *(const U4*)&Bl[br1 + 48 + h2 * 8];

        acc[0] = wmma_bf16(a0, b00, acc[0]);
        acc[0] = wmma_bf16(a1, b01, acc[0]);
        acc[1] = wmma_bf16(a0, b10, acc[1]);
        acc[1] = wmma_bf16(a1, b11, acc[1]);
        __syncthreads();
    }

#pragma unroll
    for (int pass = 0; pass < 2; ++pass) {
        const int ni  = nh + pass * 2;
        const int col = n0 + ni * 16 + m_l;
#pragma unroll
        for (int j = 0; j < 8; ++j) {
            const int row = m0 + mi * 16 + 8 * h2 + j;
            const float val = acc[pass][j];
            if (mode) Cf[(size_t)row * E_DIM + col] = val;
            else      Cb[(size_t)row * E_DIM + col] = f2bf(val);
        }
    }
}

// ---------- GEMM: out[m][n] = sum_k Abf[m][k]*Wbf[n][k] + bias[n], f32 out
__global__ __launch_bounds__(256) void k_gemm_a16_bias(
    const u16* __restrict__ A, const u16* __restrict__ Wb,
    const float* __restrict__ bias, float* __restrict__ Cf)
{
    __shared__ alignas(16) u16 Al[64 * 72];
    __shared__ alignas(16) u16 Bl[64 * 72];

    const int t    = threadIdx.x;
    const int wv_  = t >> 5, lane = t & 31;
    const int mi   = wv_ & 3, nh = wv_ >> 2;
    const int m_l  = lane & 15, h2 = lane >> 4;
    const int m0   = blockIdx.y * 64, n0 = blockIdx.x * 64;
    const int ar   = t >> 2;
    const int ac   = (t & 3) * 16;

    v8f acc[2];
    acc[0] = {}; acc[1] = {};

    for (int k = 0; k < E_DIM; k += 64) {
        const u16* as = A + (size_t)(m0 + ar) * E_DIM + k + ac;
        const u16* bs = Wb + (size_t)(n0 + ar) * E_DIM + k + ac;
        stage16(as,     &Al[ar * 72 + ac]);
        stage16(as + 8, &Al[ar * 72 + ac + 8]);
        stage16(bs,     &Bl[ar * 72 + ac]);
        stage16(bs + 8, &Bl[ar * 72 + ac + 8]);
        stage_wait();
        __syncthreads();

        Frag a0, a1;
        const int arow = (mi * 16 + m_l) * 72;
        a0.q[0] = *(const U4*)&Al[arow + h2 * 8];
        a0.q[1] = *(const U4*)&Al[arow + 16 + h2 * 8];
        a1.q[0] = *(const U4*)&Al[arow + 32 + h2 * 8];
        a1.q[1] = *(const U4*)&Al[arow + 48 + h2 * 8];

        const int br0 = (nh * 16 + m_l) * 72;
        const int br1 = ((nh + 2) * 16 + m_l) * 72;
        Frag b00, b01, b10, b11;
        b00.q[0] = *(const U4*)&Bl[br0 + h2 * 8];
        b00.q[1] = *(const U4*)&Bl[br0 + 16 + h2 * 8];
        b01.q[0] = *(const U4*)&Bl[br0 + 32 + h2 * 8];
        b01.q[1] = *(const U4*)&Bl[br0 + 48 + h2 * 8];
        b10.q[0] = *(const U4*)&Bl[br1 + h2 * 8];
        b10.q[1] = *(const U4*)&Bl[br1 + 16 + h2 * 8];
        b11.q[0] = *(const U4*)&Bl[br1 + 32 + h2 * 8];
        b11.q[1] = *(const U4*)&Bl[br1 + 48 + h2 * 8];

        acc[0] = wmma_bf16(a0, b00, acc[0]);
        acc[0] = wmma_bf16(a1, b01, acc[0]);
        acc[1] = wmma_bf16(a0, b10, acc[1]);
        acc[1] = wmma_bf16(a1, b11, acc[1]);
        __syncthreads();
    }

#pragma unroll
    for (int pass = 0; pass < 2; ++pass) {
        const int ni  = nh + pass * 2;
        const int col = n0 + ni * 16 + m_l;
        const float b = bias[col];
#pragma unroll
        for (int j = 0; j < 8; ++j) {
            const int row = m0 + mi * 16 + 8 * h2 + j;
            Cf[(size_t)row * E_DIM + col] = acc[pass][j] + b;
        }
    }
}

// ---------- fused energy (WMMA) + top-64 + softmax + A*V ----------
// block = 256 threads (8 waves); one block per (n, h, 16-query tile)
__global__ __launch_bounds__(256) void k_attn_topm(
    const u16* __restrict__ Qp, const u16* __restrict__ Kp,
    const float* __restrict__ Vp, u16* __restrict__ Ctx)
{
    __shared__ alignas(16) u16  Aq[16 * 72];
    __shared__ alignas(16) u16  Bk[128 * 72];
    __shared__ float Etile[16 * 136];
    __shared__ float topv[16][M_TOP];
    __shared__ int   topi[16][M_TOP];

    const int b  = blockIdx.x;
    const int qt = b & 127;
    const int h  = (b >> 7) & (H_NUM - 1);
    const int n  = b >> 11;
    const int q0 = qt * 16;

    const int t    = threadIdx.x;
    const int wv_  = t >> 5, lane = t & 31;
    const int m_l  = lane & 15, h2 = lane >> 4;

    // stage 16x64 query tile (bf16): 128 x 16B transfers
    if (t < 128) {
        const int r = t >> 3, c = (t & 7) * 8;
        const u16* src = Qp + (size_t)(n * S_LEN + q0 + r) * E_DIM + h * D_DIM + c;
        stage16(src, &Aq[r * 72 + c]);
    }
    stage_wait();
    __syncthreads();

    // A fragments live in registers for the whole kernel
    Frag a0, a1;
    a0.q[0] = *(const U4*)&Aq[m_l * 72 + h2 * 8];
    a0.q[1] = *(const U4*)&Aq[m_l * 72 + 16 + h2 * 8];
    a1.q[0] = *(const U4*)&Aq[m_l * 72 + 32 + h2 * 8];
    a1.q[1] = *(const U4*)&Aq[m_l * 72 + 48 + h2 * 8];

    float rmin = 0.0f;
    int   rarg = 0;

    for (int kc = 0; kc < 16; ++kc) {
        const int k0 = kc * 128;
        // stage 128x64 key chunk: 1024 x 16B transfers, 4 per thread
#pragma unroll
        for (int i = 0; i < 4; ++i) {
            const int idx = t + i * 256;
            const int r = idx >> 3, c = (idx & 7) * 8;
            const u16* src = Kp + (size_t)(n * S_LEN + k0 + r) * E_DIM + h * D_DIM + c;
            stage16(src, &Bk[r * 72 + c]);
        }
        stage_wait();
        __syncthreads();

        // wave wv_ computes energy subtile [16 q][16 k] at k-cols wv_*16..+15
        {
            const int br = (wv_ * 16 + m_l) * 72;
            Frag b0, b1;
            b0.q[0] = *(const U4*)&Bk[br + h2 * 8];
            b0.q[1] = *(const U4*)&Bk[br + 16 + h2 * 8];
            b1.q[0] = *(const U4*)&Bk[br + 32 + h2 * 8];
            b1.q[1] = *(const U4*)&Bk[br + 48 + h2 * 8];
            v8f e = {};
            e = wmma_bf16(a0, b0, e);
            e = wmma_bf16(a1, b1, e);
#pragma unroll
            for (int j = 0; j < 8; ++j)
                Etile[(8 * h2 + j) * 136 + wv_ * 16 + m_l] = e[j];
        }
        __syncthreads();

        // merge this chunk's 128 candidates into the running top-64 (1 thread/row)
        if (t < 16) {
            float* tv = &topv[t][0];
            int*   ti = &topi[t][0];
            const float* er = &Etile[t * 136];
            int cstart;
            if (kc == 0) {
                for (int j = 0; j < M_TOP; ++j) { tv[j] = er[j]; ti[j] = j; }
                rmin = tv[0]; rarg = 0;
                for (int j = 1; j < M_TOP; ++j)
                    if (tv[j] < rmin) { rmin = tv[j]; rarg = j; }
                cstart = M_TOP;
            } else {
                cstart = 0;
            }
            for (int c = cstart; c < 128; ++c) {
                const float v = er[c];
                if (v > rmin) {
                    tv[rarg] = v; ti[rarg] = k0 + c;
                    rmin = tv[0]; rarg = 0;
                    for (int j = 1; j < M_TOP; ++j)
                        if (tv[j] < rmin) { rmin = tv[j]; rarg = j; }
                }
            }
        }
        __syncthreads();
    }

    // softmax over top-64 per row (in place)
    if (t < 16) {
        float* tv = &topv[t][0];
        float mx = tv[0];
        for (int j = 1; j < M_TOP; ++j) mx = fmaxf(mx, tv[j]);
        float s = 0.0f;
        for (int j = 0; j < M_TOP; ++j) { float e = __expf(tv[j] - mx); tv[j] = e; s += e; }
        const float inv = 1.0f / s;
        for (int j = 0; j < M_TOP; ++j) tv[j] *= inv;
    }
    __syncthreads();

    // weighted gather-sum over V: each wave handles 2 q rows; lane owns d, d+32
#pragma unroll
    for (int rr = 0; rr < 2; ++rr) {
        const int r = wv_ * 2 + rr;
        float acc0 = 0.0f, acc1 = 0.0f;
        for (int m = 0; m < M_TOP; ++m) {
            const float a = topv[r][m];
            const int  ki = topi[r][m];
            const float* vp = Vp + (size_t)(n * S_LEN + ki) * E_DIM + h * D_DIM;
            acc0 += a * vp[lane];
            acc1 += a * vp[lane + 32];
        }
        const int rowg = n * S_LEN + q0 + r;
        Ctx[(size_t)rowg * E_DIM + h * D_DIM + lane]      = f2bf(acc0);
        Ctx[(size_t)rowg * E_DIM + h * D_DIM + lane + 32] = f2bf(acc1);
    }
}

// ---------- workspace layout (bytes) ----------
static const size_t OFF_WV  = 0;
static const size_t OFF_WK  = OFF_WV + (size_t)E_DIM * E_DIM * 2;
static const size_t OFF_WQ  = OFF_WK + (size_t)E_DIM * E_DIM * 2;
static const size_t OFF_WO  = OFF_WQ + (size_t)E_DIM * E_DIM * 2;
static const size_t OFF_QP  = OFF_WO + (size_t)E_DIM * E_DIM * 2;
static const size_t OFF_KP  = OFF_QP + (size_t)R_ROWS * E_DIM * 2;
static const size_t OFF_VP  = OFF_KP + (size_t)R_ROWS * E_DIM * 2;
static const size_t OFF_CTX = OFF_VP + (size_t)R_ROWS * E_DIM * 4;

extern "C" void kernel_launch(void* const* d_in, const int* in_sizes, int n_in,
                              void* d_out, int out_size, void* d_ws, size_t ws_size,
                              hipStream_t stream) {
    const float* value = (const float*)d_in[0];
    const float* key   = (const float*)d_in[1];
    const float* query = (const float*)d_in[2];
    const float* Wv    = (const float*)d_in[3];
    const float* Wk    = (const float*)d_in[4];
    const float* Wq    = (const float*)d_in[5];
    const float* Wo    = (const float*)d_in[6];
    const float* bo    = (const float*)d_in[7];
    float* out = (float*)d_out;

    char* ws  = (char*)d_ws;
    u16*  WvB = (u16*)(ws + OFF_WV);
    u16*  WkB = (u16*)(ws + OFF_WK);
    u16*  WqB = (u16*)(ws + OFF_WQ);
    u16*  WoB = (u16*)(ws + OFF_WO);
    u16*  QpB = (u16*)(ws + OFF_QP);
    u16*  KpB = (u16*)(ws + OFF_KP);
    float* Vpf = (float*)(ws + OFF_VP);
    u16*  CtxB = (u16*)(ws + OFF_CTX);

    k_cvt_weights<<<1024, 256, 0, stream>>>(Wv, Wk, Wq, Wo, WvB, WkB, WqB, WoB);

    dim3 g(E_DIM / 64, R_ROWS / 64);  // (16, 64)
    k_gemm_a32<<<g, 256, 0, stream>>>(query, WqB, nullptr, QpB, 0);
    k_gemm_a32<<<g, 256, 0, stream>>>(key,   WkB, nullptr, KpB, 0);
    k_gemm_a32<<<g, 256, 0, stream>>>(value, WvB, Vpf, nullptr, 1);

    k_attn_topm<<<NB * H_NUM * (S_LEN / 16), 256, 0, stream>>>(QpB, KpB, Vpf, CtxB);

    k_gemm_a16_bias<<<g, 256, 0, stream>>>(CtxB, WoB, bo, out);
}
